// Decoder_81209241633487
// MI455X (gfx1250) — matrix operand order repaired and checked by reading, not compiled, not run on hardware
//
#include <hip/hip_runtime.h>
#include <hip/hip_bf16.h>
#include <math.h>

// Problem dims (fixed by the reference)
#define B_ 32
#define L_ 196
#define D_ 512
#define H_ 512
#define E_ 256
#define T_ 128
#define V_ 512
#define HZE_ 1280            // H + D + E
#define NO_ 368              // 358 padded to 16-multiple (output MLP N)
#define KO_ 384              // 358 padded to 32-multiple (output MLP K)

typedef __bf16 bf16;
typedef __attribute__((ext_vector_type(16))) __bf16 v16bf;
typedef __attribute__((ext_vector_type(8)))  __bf16 v8bf;
typedef __attribute__((ext_vector_type(8)))  float  v8f;

__device__ __forceinline__ float sigm(float x) { return 1.0f / (1.0f + __expf(-x)); }

// ---- WMMA 16x16x32 bf16 fragments via 128-bit loads ----
// A (16x32, row-major source): lane holds row (lane&15); elements are
// K = half*16 + 0..15  (half = lane>>4).
__device__ __forceinline__ v16bf ld_frag_a(const bf16* p /* row base + k0 + half*16 */) {
  v8bf lo = *(const v8bf*)(p);
  v8bf hi = *(const v8bf*)(p + 8);
  return __builtin_shufflevector(lo, hi, 0, 1, 2, 3, 4, 5, 6, 7,
                                 8, 9, 10, 11, 12, 13, 14, 15);
}
// B (32x16): lane holds column (lane&15); elements are K = half*16 + 0..15.
// With B stored TRANSPOSED (N x K row-major) that's 16 contiguous bf16.
__device__ __forceinline__ v16bf ld_frag_b(const bf16* p /* BT row + k0 + half*16 */) {
  v8bf lo = *(const v8bf*)(p);
  v8bf hi = *(const v8bf*)(p + 8);
  return __builtin_shufflevector(lo, hi, 0, 1, 2, 3, 4, 5, 6, 7,
                                 8, 9, 10, 11, 12, 13, 14, 15);
}
#define WMMA_BF16(fa, fb, acc) \
  __builtin_amdgcn_wmma_f32_16x16x32_bf16(false, (fa), false, (fb), (short)0, (acc), false, false)

// =====================================================================
// Generic bf16 WMMA GEMM:  out = [tanh](A @ B + bias),  B given as BT (NxK).
// Block: 256 threads = 8 waves; one 16-row M-tile per block, TWO 16-col
// N-tiles per wave (A fragment reused -> 3 b128 loads per wmma), software
// pipelined so the wmma issues against older loads. The odd-last-tile case
// is handled by pointer aliasing (redundant load of tile0's row block)
// instead of a per-iteration branch. M % 16 == 0 always.
// =====================================================================
__global__ void gemm_bf16_kernel(const bf16* __restrict__ A, int lda,
                                 const bf16* __restrict__ BT, int ldk,
                                 const float* __restrict__ bias, int biasN,
                                 float* __restrict__ outF, bf16* __restrict__ outB,
                                 int ldo, int M, int N, int K, int doTanh) {
  const int tid = threadIdx.x, lane = tid & 31, wid = tid >> 5;
  const int ntiles = (N + 15) >> 4;
  const int nt0 = (blockIdx.x * 8 + wid) * 2;
  if (nt0 >= ntiles) return;                      // wave-uniform
  const bool has2 = (nt0 + 1) < ntiles;
  const int m0 = blockIdx.y << 4;
  const int half = lane >> 4;
  const int arow = m0 + (lane & 15);
  const int bcol0 = (nt0 << 4) + (lane & 15);
  const bf16* aptr  = A + (size_t)arow * lda + half * 16;
  const bf16* bptr0 = BT + (size_t)bcol0 * ldk + half * 16;
  const bf16* bptr1 = bptr0 + (has2 ? (size_t)16 * ldk : (size_t)0); // alias when !has2
  v8f acc0 = {}, acc1 = {};
  v16bf fa  = ld_frag_a(aptr);
  v16bf fb0 = ld_frag_b(bptr0);
  v16bf fb1 = ld_frag_b(bptr1);
#pragma unroll 2
  for (int kk = 32; kk < K; kk += 32) {
    v16bf na  = ld_frag_a(aptr + kk);
    v16bf nb0 = ld_frag_b(bptr0 + kk);
    v16bf nb1 = ld_frag_b(bptr1 + kk);
    __builtin_prefetch(bptr0 + kk + 224, 0, 1);   // speculative: OOB is dropped
    acc0 = WMMA_BF16(fa, fb0, acc0);
    acc1 = WMMA_BF16(fa, fb1, acc1);
    fa = na; fb0 = nb0; fb1 = nb1;
  }
  acc0 = WMMA_BF16(fa, fb0, acc0);
  acc1 = WMMA_BF16(fa, fb1, acc1);

  float bv0 = (bias && bcol0 < biasN) ? bias[bcol0] : 0.f;
  float bv1 = (bias && has2 && (bcol0 + 16) < biasN) ? bias[bcol0 + 16] : 0.f;
#pragma unroll
  for (int r = 0; r < 8; ++r) {
    int row = m0 + r + (half << 3);
    float v0 = acc0[r] + bv0;
    if (doTanh) v0 = tanhf(v0);
    if (outB) outB[(size_t)row * ldo + bcol0] = (bf16)v0;
    else      outF[(size_t)row * ldo + bcol0] = v0;
    if (has2) {
      float v1 = acc1[r] + bv1;
      if (doTanh) v1 = tanhf(v1);
      if (outB) outB[(size_t)row * ldo + bcol0 + 16] = (bf16)v1;
      else      outF[(size_t)row * ldo + bcol0 + 16] = v1;
    }
  }
}

// =====================================================================
// Per-step attention: one block per batch element.
// Hp = h@W1h + b1 (VALU, tiny); x1 = tanh(A1pre + Hp) staged bf16 in LDS;
// x2 = tanh(x1 @ W2 + b2) via WMMA (K=256 fully unrolled -> 8 wmma);
// scores = x2@w3 + b3; softmax; z.
// =====================================================================
__global__ void att_step_kernel(int t,
                                const float* __restrict__ a,       // [B,L,D]
                                const float* __restrict__ att_w1,  // [(D+H),256]
                                const float* __restrict__ att_b1,  // [256]
                                const float* __restrict__ att_b2,  // [128]
                                const float* __restrict__ att_w3,  // [128]
                                const float* __restrict__ att_b3,  // [1]
                                const float* __restrict__ A1pre,   // [B*L,256] (no bias)
                                const bf16*  __restrict__ w2T,     // [128,256] transposed
                                const float* __restrict__ hF,      // [B,H]
                                bf16* __restrict__ hze) {          // [B*T,1280]
  __shared__ float s_h[H_];
  __shared__ float s_Hp[256];
  __shared__ float s_sc[208];
  __shared__ __attribute__((aligned(16))) bf16 s_x1[16 * 256];
  __shared__ float s_red[256];
  const int b = blockIdx.x;
  const int tid = threadIdx.x, lane = tid & 31, wid = tid >> 5;
  const int half = lane >> 4;

  for (int i = tid; i < H_; i += 256) s_h[i] = hF[b * H_ + i];
  __syncthreads();
  { // Hp[tid] = b1 + sum_k h[k]*W1[D+k, tid]
    float acc = att_b1[tid];
    for (int k = 0; k < H_; ++k) acc += s_h[k] * att_w1[(size_t)(D_ + k) * 256 + tid];
    s_Hp[tid] = acc;
  }
  if (tid < 208) s_sc[tid] = att_b3[0];
  __syncthreads();

  const bf16* s_arow = s_x1 + (lane & 15) * 256 + half * 16;
  const bf16* bptr = w2T + (size_t)((wid << 4) + (lane & 15)) * 256 + half * 16;

  for (int tile = 0; tile < 13; ++tile) {      // 13*16 = 208 >= L=196
#pragma unroll 4
    for (int p = 0; p < 16; ++p) {             // stage tanh(A1+Hp) tile as bf16
      int idx = p * 256 + tid;
      int row = idx >> 8, col = idx & 255;
      int l = tile * 16 + row;
      float v = 0.f;
      if (l < L_) v = tanhf(A1pre[(size_t)(b * L_ + l) * 256 + col] + s_Hp[col]);
      s_x1[row * 256 + col] = (bf16)v;
    }
    __syncthreads();
    { // each wave: one 16-col tile of x2, K=256 -> 8 wmma, ds_load_b128 A frags
      v8f acc = {};
#pragma unroll
      for (int kk = 0; kk < 256; kk += 32) {
        v16bf fa = ld_frag_a(s_arow + kk);
        v16bf fb = ld_frag_b(bptr + kk);
        acc = WMMA_BF16(fa, fb, acc);
      }
      int col = (wid << 4) + (lane & 15);
      float w3v = att_w3[col], b2v = att_b2[col];
#pragma unroll
      for (int r = 0; r < 8; ++r) {
        int row = tile * 16 + r + (half << 3);
        atomicAdd(&s_sc[row], tanhf(acc[r] + b2v) * w3v);
      }
    }
    __syncthreads();
  }
  if (tid < 208 && tid >= L_) s_sc[tid] = -1e30f;
  __syncthreads();
  // softmax over L
  s_red[tid] = (tid < L_) ? s_sc[tid] : -1e30f;
  __syncthreads();
  for (int s = 128; s > 0; s >>= 1) {
    if (tid < s) s_red[tid] = fmaxf(s_red[tid], s_red[tid + s]);
    __syncthreads();
  }
  float mx = s_red[0];
  __syncthreads();
  float ev = 0.f;
  if (tid < L_) { ev = __expf(s_sc[tid] - mx); s_sc[tid] = ev; }
  s_red[tid] = ev;
  __syncthreads();
  for (int s = 128; s > 0; s >>= 1) {
    if (tid < s) s_red[tid] += s_red[tid + s];
    __syncthreads();
  }
  float inv = 1.0f / s_red[0];
  __syncthreads();
  // z[c] = (1/sum) * sum_l alpha_l * a[b,l,c]  -> bf16 into hze[.., H:H+D]
  for (int c = tid; c < D_; c += 256) {
    float acc = 0.f;
    for (int l = 0; l < L_; ++l) acc += s_sc[l] * a[(size_t)(b * L_ + l) * D_ + c];
    hze[(size_t)(b * T_ + t) * HZE_ + H_ + c] = (bf16)(acc * inv);
  }
}

// =====================================================================
// LSTM gates: gates = [z,e]@w_ih + h@w_hh + b_ih + b_hh
// M=32 (2 tiles) x N=2048 (128 tiles) = 256 wave-tiles = 32 blocks x 8 waves
// Weights pre-transposed; both K loops software pipelined.
// =====================================================================
__global__ void lstm_gates_kernel(int t,
                                  const bf16* __restrict__ hze,    // [B*T,1280]
                                  const bf16* __restrict__ hB,     // [B,512]
                                  const bf16* __restrict__ wihT,   // [2048,768]
                                  const bf16* __restrict__ whhT,   // [2048,512]
                                  const float* __restrict__ b_ih,
                                  const float* __restrict__ b_hh,
                                  float* __restrict__ gates) {     // [B,2048]
  const int tid = threadIdx.x, lane = tid & 31, wid = tid >> 5;
  const int gw = blockIdx.x * 8 + wid;
  const int m0 = (gw >> 7) << 4;
  const int n0 = (gw & 127) << 4;
  const int half = lane >> 4;
  const int brow = m0 + (lane & 15);
  const int bc = n0 + (lane & 15);
  const bf16* aptr1 = hze + (size_t)(brow * T_ + t) * HZE_ + H_ + half * 16; // [z|e]
  const bf16* bptr1 = wihT + (size_t)bc * (D_ + E_) + half * 16;
  const bf16* aptr2 = hB + (size_t)brow * H_ + half * 16;
  const bf16* bptr2 = whhT + (size_t)bc * H_ + half * 16;
  v8f acc = {};
  v16bf fa = ld_frag_a(aptr1);
  v16bf fb = ld_frag_b(bptr1);
#pragma unroll 4
  for (int kk = 32; kk < D_ + E_; kk += 32) {                 // K=768
    v16bf na = ld_frag_a(aptr1 + kk);
    v16bf nb = ld_frag_b(bptr1 + kk);
    acc = WMMA_BF16(fa, fb, acc);
    fa = na; fb = nb;
  }
  acc = WMMA_BF16(fa, fb, acc);
  fa = ld_frag_a(aptr2);
  fb = ld_frag_b(bptr2);
#pragma unroll 4
  for (int kk = 32; kk < H_; kk += 32) {                      // K=512
    v16bf na = ld_frag_a(aptr2 + kk);
    v16bf nb = ld_frag_b(bptr2 + kk);
    acc = WMMA_BF16(fa, fb, acc);
    fa = na; fb = nb;
  }
  acc = WMMA_BF16(fa, fb, acc);
  float bias = b_ih[bc] + b_hh[bc];
#pragma unroll
  for (int r = 0; r < 8; ++r) {
    int row = m0 + r + (half << 3);
    gates[(size_t)row * 2048 + bc] = acc[r] + bias;
  }
}

// LSTM nonlinearity + state update; writes h into hze[.., 0:H]
__global__ void lstm_update_kernel(int t, const float* __restrict__ gates,
                                   float* __restrict__ hF, float* __restrict__ cF,
                                   bf16* __restrict__ hB, bf16* __restrict__ hze) {
  int idx = blockIdx.x * 256 + threadIdx.x;        // B*H = 16384
  if (idx >= B_ * H_) return;
  int b = idx >> 9, j = idx & 511;
  const float* g = gates + (size_t)b * 2048;
  float i = sigm(g[j]);
  float f = sigm(g[512 + j]);
  float gg = tanhf(g[1024 + j]);
  float o = sigm(g[1536 + j]);
  float c = f * cF[idx] + i * gg;
  float h = o * tanhf(c);
  cF[idx] = c;
  hF[idx] = h;
  hB[idx] = (bf16)h;
  hze[(size_t)(b * T_ + t) * HZE_ + j] = (bf16)h;
}

// ---- prep kernels ----
__global__ void cvt_kernel(const float* __restrict__ src, bf16* __restrict__ dst, int n) {
  int idx = blockIdx.x * 256 + threadIdx.x;
  if (idx < n) dst[idx] = (bf16)src[idx];
}
// convert + TRANSPOSE + zero-pad: src is [srK x srN] row-major,
// dst is [dN x dK] row-major (i.e. B^T), padded with zeros.
__global__ void cvt_t_pad_kernel(const float* __restrict__ src, int srK, int srN,
                                 bf16* __restrict__ dst, int dN, int dK) {
  int idx = blockIdx.x * 256 + threadIdx.x;
  if (idx >= dN * dK) return;
  int n = idx / dK, k = idx - n * dK;
  float v = (k < srK && n < srN) ? src[(size_t)k * srN + n] : 0.f;
  dst[idx] = (bf16)v;
}

__global__ void hc_init_kernel(const float* __restrict__ h0, const float* __restrict__ c0,
                               float* __restrict__ hF, float* __restrict__ cF,
                               bf16* __restrict__ hB) {
  int idx = blockIdx.x * 256 + threadIdx.x;
  if (idx >= B_ * H_) return;
  float h = h0[idx];
  hF[idx] = h; cF[idx] = c0[idx]; hB[idx] = (bf16)h;
}

// embeddings of the right-shifted token sequence, straight into hze[.., H+D:]
__global__ void embed_gather_kernel(const int* __restrict__ y, const float* __restrict__ embed,
                                    bf16* __restrict__ hze) {
  int idx = blockIdx.x * 256 + threadIdx.x;        // B*T*E
  if (idx >= B_ * T_ * E_) return;
  int j = idx & (E_ - 1);
  int bt = idx >> 8;                               // E_ = 256
  int b = bt / T_, t = bt - b * T_;
  int tok = (t == 0) ? 0 : y[b * T_ + t - 1];
  hze[(size_t)bt * HZE_ + H_ + D_ + j] = (bf16)embed[(size_t)tok * E_ + j];
}

__global__ void zero_cols_kernel(bf16* __restrict__ X, int rows, int ld, int c0, int c1) {
  int w = c1 - c0;
  int idx = blockIdx.x * 256 + threadIdx.x;
  if (idx >= rows * w) return;
  int r = idx / w, c = c0 + (idx - r * w);
  X[(size_t)r * ld + c] = (bf16)0.f;
}

// =====================================================================
extern "C" void kernel_launch(void* const* d_in, const int* in_sizes, int n_in,
                              void* d_out, int out_size, void* d_ws, size_t ws_size,
                              hipStream_t stream) {
  const float* a      = (const float*)d_in[0];
  const float* h0     = (const float*)d_in[1];
  const float* c0     = (const float*)d_in[2];
  const int*   y      = (const int*)  d_in[3];
  const float* att_w1 = (const float*)d_in[4];
  const float* att_b1 = (const float*)d_in[5];
  const float* att_w2 = (const float*)d_in[6];
  const float* att_b2 = (const float*)d_in[7];
  const float* att_w3 = (const float*)d_in[8];
  const float* att_b3 = (const float*)d_in[9];
  const float* w_ih   = (const float*)d_in[10];
  const float* w_hh   = (const float*)d_in[11];
  const float* b_ih   = (const float*)d_in[12];
  const float* b_hh   = (const float*)d_in[13];
  const float* embed  = (const float*)d_in[14];
  const float* out_w1 = (const float*)d_in[15];
  const float* out_b1 = (const float*)d_in[16];
  const float* out_w2 = (const float*)d_in[17];
  const float* out_b2 = (const float*)d_in[18];
  const float* out_w3 = (const float*)d_in[19];
  const float* out_b3 = (const float*)d_in[20];
  float* logits = (float*)d_out;                   // [B*T, V]

  char* ws = (char*)d_ws;
  size_t off = 0;
  auto alloc = [&](size_t bytes) -> char* {
    char* p = ws + off;
    off = (off + bytes + 255) & ~(size_t)255;
    return p;
  };
  bf16* a_bf   = (bf16*)alloc((size_t)B_ * L_ * D_ * 2);      // A, row-major
  bf16* w1aT   = (bf16*)alloc((size_t)256 * D_ * 2);          // [256][512]
  bf16* w2T    = (bf16*)alloc((size_t)128 * 256 * 2);         // [128][256]
  bf16* wihT   = (bf16*)alloc((size_t)2048 * (D_ + E_) * 2);  // [2048][768]
  bf16* whhT   = (bf16*)alloc((size_t)2048 * H_ * 2);         // [2048][512]
  bf16* ow1T   = (bf16*)alloc((size_t)NO_ * HZE_ * 2);        // [368][1280]
  bf16* ow2T   = (bf16*)alloc((size_t)NO_ * KO_ * 2);         // [368][384]
  bf16* ow3T   = (bf16*)alloc((size_t)V_ * KO_ * 2);          // [512][384]
  float* A1pre = (float*)alloc((size_t)B_ * L_ * 256 * 4);    // 6.4 MB
  bf16* hze    = (bf16*)alloc((size_t)B_ * T_ * HZE_ * 2);    // 10.5 MB
  bf16* X1     = (bf16*)alloc((size_t)B_ * T_ * KO_ * 2);
  bf16* X2     = (bf16*)alloc((size_t)B_ * T_ * KO_ * 2);
  float* gates = (float*)alloc((size_t)B_ * 2048 * 4);
  float* hF    = (float*)alloc((size_t)B_ * H_ * 4);
  float* cF    = (float*)alloc((size_t)B_ * H_ * 4);
  bf16* hB     = (bf16*)alloc((size_t)B_ * H_ * 2);

  dim3 blk(256);
  auto g1 = [](size_t n) { return dim3((unsigned)((n + 255) / 256)); };
  // waves do 2 N-tiles each; 8 waves per block
  auto gx = [](int N) { return (unsigned)(((((N + 15) / 16 + 1) / 2) + 7) / 8); };

  // --- prep: bf16 conversion of A operand; transpose+pad all B operands ---
  cvt_kernel<<<g1((size_t)B_ * L_ * D_), blk, 0, stream>>>(a, a_bf, B_ * L_ * D_);
  cvt_t_pad_kernel<<<g1((size_t)256 * D_), blk, 0, stream>>>(att_w1, D_, 256, w1aT, 256, D_);
  cvt_t_pad_kernel<<<g1((size_t)128 * 256), blk, 0, stream>>>(att_w2, 256, 128, w2T, 128, 256);
  cvt_t_pad_kernel<<<g1((size_t)2048 * (D_ + E_)), blk, 0, stream>>>(w_ih, D_ + E_, 2048, wihT, 2048, D_ + E_);
  cvt_t_pad_kernel<<<g1((size_t)2048 * H_), blk, 0, stream>>>(w_hh, H_, 2048, whhT, 2048, H_);
  cvt_t_pad_kernel<<<g1((size_t)NO_ * HZE_), blk, 0, stream>>>(out_w1, HZE_, 358, ow1T, NO_, HZE_);
  cvt_t_pad_kernel<<<g1((size_t)NO_ * KO_), blk, 0, stream>>>(out_w2, 358, 358, ow2T, NO_, KO_);
  cvt_t_pad_kernel<<<g1((size_t)V_ * KO_), blk, 0, stream>>>(out_w3, 358, V_, ow3T, V_, KO_);
  hc_init_kernel<<<g1(B_ * H_), blk, 0, stream>>>(h0, c0, hF, cF, hB);
  embed_gather_kernel<<<g1((size_t)B_ * T_ * E_), blk, 0, stream>>>(y, embed, hze);
  zero_cols_kernel<<<g1((size_t)B_ * T_ * (KO_ - NO_)), blk, 0, stream>>>(X1, B_ * T_, KO_, NO_, KO_);
  zero_cols_kernel<<<g1((size_t)B_ * T_ * (KO_ - NO_)), blk, 0, stream>>>(X2, B_ * T_, KO_, NO_, KO_);

  // --- hoisted attention precompute: A1pre = a @ att_w1[:D]  (bias added in Hp) ---
  gemm_bf16_kernel<<<dim3(gx(256), (B_ * L_) / 16), blk, 0, stream>>>(
      a_bf, D_, w1aT, D_, nullptr, 0, A1pre, nullptr, 256,
      B_ * L_, 256, D_, 0);

  // --- sequential scan over T (stream ordering carries the h/c dependency) ---
  for (int t = 0; t < T_; ++t) {
    att_step_kernel<<<dim3(B_), blk, 0, stream>>>(
        t, a, att_w1, att_b1, att_b2, att_w3, att_b3, A1pre, w2T, hF, hze);
    lstm_gates_kernel<<<dim3(32), blk, 0, stream>>>(
        t, hze, hB, wihT, whhT, b_ih, b_hh, gates);
    lstm_update_kernel<<<g1(B_ * H_), blk, 0, stream>>>(t, gates, hF, cF, hB, hze);
  }

  // --- output MLP: tanh GEMMs then logits ---
  gemm_bf16_kernel<<<dim3(gx(NO_), (B_ * T_) / 16), blk, 0, stream>>>(
      hze, HZE_, ow1T, HZE_, out_b1, 358, nullptr, X1, KO_,
      B_ * T_, NO_, HZE_, 1);
  gemm_bf16_kernel<<<dim3(gx(NO_), (B_ * T_) / 16), blk, 0, stream>>>(
      X1, KO_, ow2T, KO_, out_b2, 358, nullptr, X2, KO_,
      B_ * T_, NO_, KO_, 1);
  gemm_bf16_kernel<<<dim3(gx(V_), (B_ * T_) / 16), blk, 0, stream>>>(
      X2, KO_, ow3T, KO_, out_b3, V_, logits, nullptr, V_,
      B_ * T_, V_, KO_, 0);
}
